// SubMConv3dTorch_38491496906942
// MI455X (gfx1250) — matrix operand order repaired and checked by reading, not compiled, not run on hardware
//
#include <hip/hip_runtime.h>
#include <hip/hip_bf16.h>

// ---- problem constants (match reference) ----
#define CIN   64
#define COUT  64
#define D12   16384        // D1*D2
#define D2C   128          // D2
#define VOLC  2097152      // D0*D1*D2

// ---- hash table ----
#define HBITS 20
#define HSIZE (1 << HBITS)
#define HMASK (HSIZE - 1)

// ---- LDS weight staging: rows padded 64 -> 72 halves (bank-conflict-free) ----
#define LROW 72

typedef __attribute__((ext_vector_type(16))) _Float16 v16h;
typedef __attribute__((ext_vector_type(8)))  _Float16 v8h;
typedef __attribute__((ext_vector_type(8)))  float    v8f;

__device__ __forceinline__ uint32_t hash_slot(int key) {
  uint32_t h = (uint32_t)key * 2654435761u;
  return (h >> 11) & HMASK;
}

__global__ void k_hash_init(int* __restrict__ hk) {
  int i = blockIdx.x * blockDim.x + threadIdx.x;
  if (i < HSIZE) hk[i] = -1;
}

// weight [COUT][27][CIN] fp32  ->  wh [tap][out][in] f16
__global__ void k_wconv(const float* __restrict__ w, _Float16* __restrict__ wh) {
  int id = blockIdx.x * blockDim.x + threadIdx.x;
  if (id < 27 * COUT * CIN) {
    int i = id & 63;
    int o = (id >> 6) & 63;
    int t = id >> 12;
    wh[t * (COUT * CIN) + o * CIN + i] = (_Float16)w[o * (27 * CIN) + t * CIN + i];
  }
}

// features fp32 -> f16 (one-shot; removes all cvt from hot loop)
__global__ void k_fconv(const float* __restrict__ f, _Float16* __restrict__ fh, int total) {
  int id = blockIdx.x * blockDim.x + threadIdx.x;
  if (id < total) fh[id] = (_Float16)f[id];
}

__global__ void k_build(const int* __restrict__ idx, int n,
                        int* __restrict__ hk, int* __restrict__ hv,
                        int* __restrict__ packed) {
  int i = blockIdx.x * blockDim.x + threadIdx.x;
  if (i >= n) return;
  int b  = idx[i * 4 + 0];
  int x0 = idx[i * 4 + 1];
  int x1 = idx[i * 4 + 2];
  int x2 = idx[i * 4 + 3];
  int p = b * VOLC + x0 * D12 + x1 * D2C + x2;
  packed[i] = p;
  uint32_t h = hash_slot(p);
  while (true) {
    int old = atomicCAS(&hk[h], -1, p);
    if (old == -1 || old == p) { hv[h] = i; break; }
    h = (h + 1) & HMASK;
  }
}

__device__ __forceinline__ v16h cat16(v8h lo, v8h hi) {
  return __builtin_shufflevector(lo, hi, 0, 1, 2, 3, 4, 5, 6, 7,
                                         8, 9, 10, 11, 12, 13, 14, 15);
}

// one wave = 16-row x 64-outch tile; block = 8 waves = 128 rows.
// Tap weights double-buffered in LDS, shared by all 8 waves.
__global__ void __launch_bounds__(256)
k_conv(const _Float16* __restrict__ fh,
       const _Float16* __restrict__ wh,
       const float* __restrict__ bias,
       const int* __restrict__ hk, const int* __restrict__ hv,
       const int* __restrict__ packed,
       float* __restrict__ out, int n) {
  __shared__ _Float16 lbuf[2][COUT * LROW];

  int tid     = threadIdx.x;
  int lane    = tid & 31;
  int wave    = tid >> 5;
  int rowBase = (blockIdx.x * 8 + wave) * 16;
  bool waveActive = rowBase < n;

  int colw  = lane & 15;
  int myrow = rowBase + colw;
  bool rowok = waveActive && (myrow < n);
  int mypack = rowok ? packed[myrow] : (int)0x80000000;

  int khalfA = (lane >> 4) << 4;      // A: lanes>=16 read K+16 half
  int khalfB = (lane >> 4) << 4;      // B: lanes>=16 read K+16 half

  // staging map: each thread moves 16 halves of the 4096-half tap block
  int so = tid >> 2;                  // out channel 0..63
  int sc = (tid & 3) << 4;            // 0/16/32/48 in-channel offset
  const _Float16* wsrc = wh + so * CIN + sc;
  _Float16* sd0 = &lbuf[0][so * LROW + sc];
  _Float16* sd1 = &lbuf[1][so * LROW + sc];

  // preload tap 0 into buffer 0
  {
    v8h w0 = *(const v8h*)(wsrc);
    v8h w1 = *(const v8h*)(wsrc + 8);
    *(v8h*)(sd0)     = w0;
    *(v8h*)(sd0 + 8) = w1;
  }
  __syncthreads();

  v8f acc0 = {}, acc1 = {}, acc2 = {}, acc3 = {};

  for (int t = 0; t < 27; ++t) {
    // ---- prefetch next tap's weights into registers (overlaps compute) ----
    v8h wn0 = {}, wn1 = {};
    if (t + 1 < 27) {
      const _Float16* p = wsrc + (t + 1) * (COUT * CIN);
      wn0 = *(const v8h*)(p);
      wn1 = *(const v8h*)(p + 8);
    }

    // ---- neighbor lookup ----
    int shift = (t / 9 - 1) * D12 + ((t / 3) % 3 - 1) * D2C + (t % 3 - 1);
    int nb;
    if (!rowok) {
      nb = -1;
    } else if (shift == 0) {
      nb = myrow;                     // center tap always hits self
    } else {
      int key = mypack + shift;
      uint32_t h = hash_slot(key);
      nb = -1;
      while (true) {
        int k = hk[h];
        if (k == key) { nb = hv[h]; break; }
        if (k == -1) break;
        h = (h + 1) & HMASK;
      }
    }

    if (waveActive && __any(nb >= 0)) {   // wave-uniform; EXEC full for WMMA
      // ---- A fragments: masked 16B f16 global loads ----
      v16h a0 = {}, a1 = {};
      if (nb >= 0) {
        const _Float16* fr = fh + (size_t)nb * CIN + khalfA;
        a0 = cat16(*(const v8h*)(fr),      *(const v8h*)(fr + 8));
        a1 = cat16(*(const v8h*)(fr + 32), *(const v8h*)(fr + 40));
      }

      // ---- B fragments from LDS (conflict-free ds_load_b128) ----
      const _Float16* lb = &lbuf[t & 1][colw * LROW + khalfB];
      v16h b00 = cat16(*(const v8h*)(lb + 0 * 16 * LROW),      *(const v8h*)(lb + 0 * 16 * LROW + 8));
      v16h b01 = cat16(*(const v8h*)(lb + 0 * 16 * LROW + 32), *(const v8h*)(lb + 0 * 16 * LROW + 40));
      v16h b10 = cat16(*(const v8h*)(lb + 1 * 16 * LROW),      *(const v8h*)(lb + 1 * 16 * LROW + 8));
      v16h b11 = cat16(*(const v8h*)(lb + 1 * 16 * LROW + 32), *(const v8h*)(lb + 1 * 16 * LROW + 40));
      v16h b20 = cat16(*(const v8h*)(lb + 2 * 16 * LROW),      *(const v8h*)(lb + 2 * 16 * LROW + 8));
      v16h b21 = cat16(*(const v8h*)(lb + 2 * 16 * LROW + 32), *(const v8h*)(lb + 2 * 16 * LROW + 40));
      v16h b30 = cat16(*(const v8h*)(lb + 3 * 16 * LROW),      *(const v8h*)(lb + 3 * 16 * LROW + 8));
      v16h b31 = cat16(*(const v8h*)(lb + 3 * 16 * LROW + 32), *(const v8h*)(lb + 3 * 16 * LROW + 40));

      acc0 = __builtin_amdgcn_wmma_f32_16x16x32_f16(false, a0, false, b00, (short)0, acc0, false, false);
      acc0 = __builtin_amdgcn_wmma_f32_16x16x32_f16(false, a1, false, b01, (short)0, acc0, false, false);
      acc1 = __builtin_amdgcn_wmma_f32_16x16x32_f16(false, a0, false, b10, (short)0, acc1, false, false);
      acc1 = __builtin_amdgcn_wmma_f32_16x16x32_f16(false, a1, false, b11, (short)0, acc1, false, false);
      acc2 = __builtin_amdgcn_wmma_f32_16x16x32_f16(false, a0, false, b20, (short)0, acc2, false, false);
      acc2 = __builtin_amdgcn_wmma_f32_16x16x32_f16(false, a1, false, b21, (short)0, acc2, false, false);
      acc3 = __builtin_amdgcn_wmma_f32_16x16x32_f16(false, a0, false, b30, (short)0, acc3, false, false);
      acc3 = __builtin_amdgcn_wmma_f32_16x16x32_f16(false, a1, false, b31, (short)0, acc3, false, false);
    }

    // ---- commit prefetched weights to the other buffer ----
    if (t + 1 < 27) {
      _Float16* d = ((t + 1) & 1) ? sd1 : sd0;
      *(v8h*)(d)     = wn0;
      *(v8h*)(d + 8) = wn1;
    }
    __syncthreads();
  }

  // store D per 16x16 f32 C/D layout: lane<16 -> rows v, lane>=16 -> rows v+8
  if (waveActive) {
    int rbase = rowBase + ((lane >> 4) << 3);
#pragma unroll
    for (int ot = 0; ot < 4; ++ot) {
      v8f acc = ot == 0 ? acc0 : ot == 1 ? acc1 : ot == 2 ? acc2 : acc3;
      float bv = bias[ot * 16 + colw];
#pragma unroll
      for (int v = 0; v < 8; ++v) {
        int row = rbase + v;
        if (row < n) out[(size_t)row * COUT + ot * 16 + colw] = acc[v] + bv;
      }
    }
  }
}

extern "C" void kernel_launch(void* const* d_in, const int* in_sizes, int n_in,
                              void* d_out, int out_size, void* d_ws, size_t ws_size,
                              hipStream_t stream) {
  const float* feats  = (const float*)d_in[0];
  const float* weight = (const float*)d_in[1];
  const float* bias   = (const float*)d_in[2];
  const int*   indices= (const int*)d_in[3];
  float* out = (float*)d_out;
  int n = in_sizes[0] / CIN;

  char* ws = (char*)d_ws;
  int* hk      = (int*)ws;
  int* hv      = (int*)(ws + (size_t)HSIZE * 4);
  int* packed  = (int*)(ws + (size_t)HSIZE * 8);
  size_t whOff = (size_t)HSIZE * 8 + (((size_t)n * 4 + 31) / 32) * 32;
  _Float16* wh = (_Float16*)(ws + whOff);
  size_t fhOff = whOff + ((27 * COUT * CIN * 2 + 31) / 32) * 32;
  _Float16* fh = (_Float16*)(ws + fhOff);

  k_hash_init<<<(HSIZE + 255) / 256, 256, 0, stream>>>(hk);
  k_wconv<<<(27 * COUT * CIN + 255) / 256, 256, 0, stream>>>(weight, wh);
  k_fconv<<<(n * CIN + 255) / 256, 256, 0, stream>>>(feats, fh, n * CIN);
  k_build<<<(n + 255) / 256, 256, 0, stream>>>(indices, n, hk, hv, packed);
  k_conv<<<(n + 127) / 128, 256, 0, stream>>>(fh, wh, bias, hk, hv, packed, out, n);
}
